// LSTM_47021301957392
// MI455X (gfx1250) — matrix operand order repaired and checked by reading, not compiled, not run on hardware
//
#include <hip/hip_runtime.h>
#include <hip/hip_bf16.h>

#ifndef __has_builtin
#define __has_builtin(x) 0
#endif

#define B_   256
#define S_   512
#define H_   512
#define Z_   514
#define NG_  2048      // 4*H
#define WU_K 516

#define NBLK 64
#define NTHR 256

typedef __bf16 bf16;
typedef __bf16 v16bf __attribute__((ext_vector_type(16)));
typedef __bf16 v8bf  __attribute__((ext_vector_type(8)));
typedef float  v8f   __attribute__((ext_vector_type(8)));
typedef float  v4f   __attribute__((ext_vector_type(4)));
typedef int    v4i   __attribute__((ext_vector_type(4)));

typedef __attribute__((address_space(1))) v4i* gv4i_p;  // global
typedef __attribute__((address_space(3))) v4i* lv4i_p;  // LDS

// ---- workspace layout (bytes) ----
#define WS_BAR  0ULL
#define WS_W    64ULL                          // NG_*H_ bf16 (2 MB)
#define WS_WXU  (WS_W + (size_t)NG_ * H_ * 2)  // NG_*4 f32 (32 KB)
#define WS_HB   (WS_WXU + (size_t)NG_ * 4 * 4) // 2 * B_*H_ bf16 (512 KB)
#define HBSZ    ((size_t)B_ * H_)

// ---- output offsets (float elements) ----
#define OUT_ZF  ((size_t)B_ * S_ * Z_)         // outputs: (B,S,Z)
#define OUT_CZ  (OUT_ZF + (size_t)B_ * Z_)     // z_f: (B,Z)
#define OUT_CO  (OUT_CZ + (size_t)B_ * Z_)     // cz_f: (B,Z); then coefficients (B,S,2)

__device__ __forceinline__ float sigmoidf_(float x) { return 1.f / (1.f + __expf(-x)); }

__device__ __forceinline__ void grid_barrier(unsigned* cnt, unsigned* gen, unsigned iter) {
  __syncthreads();
  if (threadIdx.x == 0) {
    __threadfence();
    unsigned prev = __hip_atomic_fetch_add(cnt, 1u, __ATOMIC_ACQ_REL, __HIP_MEMORY_SCOPE_AGENT);
    if (prev == (unsigned)(gridDim.x - 1)) {
      __hip_atomic_store(cnt, 0u, __ATOMIC_RELAXED, __HIP_MEMORY_SCOPE_AGENT);
      __hip_atomic_store(gen, iter + 1u, __ATOMIC_RELEASE, __HIP_MEMORY_SCOPE_AGENT);
    } else {
      while (__hip_atomic_load(gen, __ATOMIC_ACQUIRE, __HIP_MEMORY_SCOPE_AGENT) < iter + 1u)
        __builtin_amdgcn_s_sleep(2);
    }
    __threadfence();
  }
  __syncthreads();
}

// Stage this block's 16x512 bf16 h-slab into LDS (1024 16-byte chunks, 256 threads x 4).
__device__ __forceinline__ void stage_slab(bf16 (*hslab)[H_ + 8], const bf16* hsrc, int tid) {
#pragma unroll
  for (int i = 0; i < 4; ++i) {
    int c = tid + i * NTHR;           // chunk 0..1023
    int r = c >> 6;                   // 64 chunks per 512-elem row
    int k = (c & 63) << 3;
#if __has_builtin(__builtin_amdgcn_global_load_async_to_lds_b128)
    __builtin_amdgcn_global_load_async_to_lds_b128(
        (gv4i_p)(hsrc + (size_t)r * H_ + k),
        (lv4i_p)&hslab[r][k], 0, 0);
#else
    *(v8bf*)&hslab[r][k] = *(const v8bf*)(hsrc + (size_t)r * H_ + k);
#endif
  }
#if __has_builtin(__builtin_amdgcn_global_load_async_to_lds_b128)
#if __has_builtin(__builtin_amdgcn_s_wait_asynccnt)
  __builtin_amdgcn_s_wait_asynccnt(0);
#else
  asm volatile("s_wait_asynccnt 0" ::: "memory");
#endif
#endif
}

// ---- prep: bf16 weight pack, x/u weight head, h0, barrier init ----
__global__ void prep_kernel(const float* __restrict__ WU_w,
                            const float* __restrict__ z0,
                            unsigned char* __restrict__ ws) {
  bf16* W = (bf16*)(ws + WS_W);
  float* Wxu = (float*)(ws + WS_WXU);
  bf16* h0 = (bf16*)(ws + WS_HB);
  unsigned* bar = (unsigned*)(ws + WS_BAR);
  const int tid = blockIdx.x * blockDim.x + threadIdx.x;
  const int nth = gridDim.x * blockDim.x;
  if (tid == 0) { bar[0] = 0u; bar[1] = 0u; }
  for (int i = tid; i < NG_ * H_; i += nth) {
    int n = i >> 9, k = i & (H_ - 1);
    W[i] = (bf16)WU_w[(size_t)n * WU_K + 4 + k];
  }
  for (int i = tid; i < NG_ * 4; i += nth) {
    int n = i >> 2, k = i & 3;
    Wxu[i] = WU_w[(size_t)n * WU_K + k];
  }
  for (int i = tid; i < B_ * H_; i += nth) {
    int b = i >> 9, k = i & (H_ - 1);
    h0[i] = (bf16)z0[(size_t)b * Z_ + 2 + k];
  }
}

// ---- persistent recurrent kernel: 64 blocks x 256 threads = 512 waves = 16x32 tiles ----
__global__ __launch_bounds__(NTHR)
void lstm_persist(const float* __restrict__ rnn_input, const float* __restrict__ tau,
                  const float* __restrict__ z0, const float* __restrict__ c0,
                  const float* __restrict__ WU_b,
                  const float* __restrict__ alpha_w, const float* __restrict__ alpha_b,
                  const float* __restrict__ Wh_w, const float* __restrict__ Wh_b,
                  const float* __restrict__ Amat, const float* __restrict__ Bmat,
                  float* __restrict__ out, unsigned char* __restrict__ ws) {
  __shared__ __align__(16) bf16 hslab[16][H_ + 8];  // +8 pad: de-conflict LDS banks
  __shared__ float xlds[16][2];

  unsigned* bar = (unsigned*)(ws + WS_BAR);
  const bf16* W = (const bf16*)(ws + WS_W);
  const float* Wxu = (const float*)(ws + WS_WXU);
  bf16* hbuf = (bf16*)(ws + WS_HB);

  const int tid  = threadIdx.x;
  const int lane = tid & 31;
  const int wib  = tid >> 5;                 // wave in block 0..7
  const int mt   = blockIdx.x >> 2;          // 16 m-tiles, shared by 4 blocks
  const int nt   = ((blockIdx.x & 3) << 3) + wib;  // 32 n-tiles
  const int b0   = mt << 4;
  const int j0   = nt << 4;
  const int l    = lane & 15;
  const int hi   = lane >> 4;
  const int rbase = hi << 3;                 // C/D layout: M = v + 8*hi, N = lane&15
  const bool writer = (blockIdx.x & 3) == 0; // one of 4 redundant x-phase copies writes

  const float a00 = Amat[0], a01 = Amat[1], a10 = Amat[2], a11 = Amat[3];
  const float bm0 = Bmat[0], bm1 = Bmat[1];
  const float ab0 = alpha_b[0], ab1 = alpha_b[1];
  const float wb0 = Wh_b[0], wb1 = Wh_b[1];

  // cell state tile lives in registers for the whole sequence
  float creg[8];
#pragma unroll
  for (int v = 0; v < 8; ++v)
    creg[v] = c0[(size_t)(b0 + rbase + v) * Z_ + 2 + j0 + l];

  if (wib == 0 && lane < 16) {
    xlds[lane][0] = z0[(size_t)(b0 + lane) * Z_ + 0];
    xlds[lane][1] = z0[(size_t)(b0 + lane) * Z_ + 1];
  }

  // alpha(tp)/x(tp) from the staged h(tp) slab; runs on wave 0 only
  auto xphase = [&](int tp, bool final_step) {
    for (int r = 0; r < 16; ++r) {
      float pa0 = 0.f, pa1 = 0.f, pw0 = 0.f, pw1 = 0.f;
      const int kb = lane << 4;
#pragma unroll 4
      for (int k = 0; k < 16; ++k) {
        float hv = (float)hslab[r][kb + k];
        pa0 += hv * alpha_w[kb + k];
        pa1 += hv * alpha_w[H_ + kb + k];
        pw0 += hv * Wh_w[kb + k];
        pw1 += hv * Wh_w[H_ + kb + k];
      }
#pragma unroll
      for (int s = 16; s > 0; s >>= 1) {
        pa0 += __shfl_xor(pa0, s, 32);
        pa1 += __shfl_xor(pa1, s, 32);
        pw0 += __shfl_xor(pw0, s, 32);
        pw1 += __shfl_xor(pw1, s, 32);
      }
      if (lane == 0) {
        const int b = b0 + r;
        float x0p = xlds[r][0], x1p = xlds[r][1];
        float tt = tau[(size_t)b * S_ + tp];
        float u0 = rnn_input[((size_t)b * S_ + tp) * 2 + 0];
        float xm0 = x0p + tt * (a00 * x0p + a01 * x1p + bm0 * u0);
        float xm1 = x1p + tt * (a10 * x0p + a11 * x1p + bm1 * u0);
        float al0 = sigmoidf_(pa0 + ab0);
        float al1 = sigmoidf_(pa1 + ab1);
        float xn0 = al0 * xm0 + (1.f - al0) * (pw0 + wb0);
        float xn1 = al1 * xm1 + (1.f - al1) * (pw1 + wb1);
        xlds[r][0] = xn0; xlds[r][1] = xn1;
        if (writer) {
          size_t ob = (size_t)b * S_ * Z_ + (size_t)tp * Z_;
          out[ob + 0] = xn0;
          out[ob + 1] = xn1;
          size_t oc = OUT_CO + (size_t)B_ * Z_ /*skip cz_f*/ + ((size_t)b * S_ + tp) * 2;
          out[oc + 0] = al0;
          out[oc + 1] = al1;
          if (final_step) {
            out[OUT_ZF + (size_t)b * Z_ + 0] = xn0;
            out[OUT_ZF + (size_t)b * Z_ + 1] = xn1;
            out[OUT_CZ + (size_t)b * Z_ + 0] = c0[(size_t)b * Z_ + 0];
            out[OUT_CZ + (size_t)b * Z_ + 1] = c0[(size_t)b * Z_ + 1];
          }
        }
      }
    }
  };

  for (int t = 0; t < S_; ++t) {
    grid_barrier(bar, bar + 1, (unsigned)t);

    const bf16* hcur = hbuf + (size_t)(t & 1) * HBSZ + (size_t)b0 * H_;
    bf16* hnxt = hbuf + (size_t)((t + 1) & 1) * HBSZ;

    stage_slab(hslab, hcur, tid);
    __syncthreads();

    if (t > 0 && wib == 0) xphase(t - 1, false);
    __syncthreads();

    // ---- gate tile: D = h(t-1) @ W_gate^T, bf16 WMMA, f32 accum ----
    v8f acc[4];
    {
      float xr0[8], xr1[8], uu0[8], uu1[8];
#pragma unroll
      for (int v = 0; v < 8; ++v) {
        const int r = rbase + v;
        const int b = b0 + r;
        xr0[v] = xlds[r][0];
        xr1[v] = xlds[r][1];
        const float* up = rnn_input + ((size_t)b * S_ + t) * 2;
        uu0[v] = up[0];
        uu1[v] = up[1];
      }
#pragma unroll
      for (int g = 0; g < 4; ++g) {
        const int nc = (g << 9) + j0 + l;
        const float bias = WU_b[nc];
        const v4f wv = *(const v4f*)(Wxu + (size_t)nc * 4);
#pragma unroll
        for (int v = 0; v < 8; ++v)
          acc[g][v] = bias + xr0[v] * wv.x + xr1[v] * wv.y + uu0[v] * wv.z + uu1[v] * wv.w;
      }
    }

#pragma unroll 4
    for (int kk = 0; kk < 16; ++kk) {
      const int k0 = kk << 5;
      // A operand (16x32 bf16): lanes 0-15 K[0:8)/[16:24), lanes 16-31 K[8:16)/[24:32)
      v8bf alo = *(const v8bf*)&hslab[l][k0 + (hi << 3)];
      v8bf ahi = *(const v8bf*)&hslab[l][k0 + 16 + (hi << 3)];
      v16bf Aop = __builtin_shufflevector(alo, ahi, 0, 1, 2, 3, 4, 5, 6, 7, 8, 9, 10, 11, 12, 13, 14, 15);
#pragma unroll
      for (int g = 0; g < 4; ++g) {
        // B operand (32x16 bf16): lane = column, lanes 0-15 K[0:8)/[16:24), lanes 16-31 K[8:16)/[24:32)
        const bf16* bp = W + (size_t)((g << 9) + j0 + l) * H_ + k0 + (hi << 3);
        v8bf blo = *(const v8bf*)bp;
        v8bf bhi = *(const v8bf*)(bp + 16);
        v16bf Bop = __builtin_shufflevector(blo, bhi, 0, 1, 2, 3, 4, 5, 6, 7, 8, 9, 10, 11, 12, 13, 14, 15);
        acc[g] = __builtin_amdgcn_wmma_f32_16x16x32_bf16(false, Aop, false, Bop, (short)0, acc[g],
                                                         false, false);
      }
    }

    // ---- cell update, h writeback ----
#pragma unroll
    for (int v = 0; v < 8; ++v) {
      const int b = b0 + rbase + v;
      float iv = sigmoidf_(acc[0][v]);
      float fv = sigmoidf_(acc[1][v]);
      float gv = tanhf(acc[2][v]);
      float ov = sigmoidf_(acc[3][v]);
      float cn = fv * creg[v] + iv * gv;
      float hn = ov * tanhf(cn);
      creg[v] = cn;
      out[(size_t)b * S_ * Z_ + (size_t)t * Z_ + 2 + j0 + l] = hn;
      hnxt[(size_t)b * H_ + j0 + l] = (bf16)hn;
      if (t == S_ - 1) out[OUT_ZF + (size_t)b * Z_ + 2 + j0 + l] = hn;
    }
  }

  // ---- epilogue: alpha(S-1)/x(S-1) and final states ----
  grid_barrier(bar, bar + 1, (unsigned)S_);
  stage_slab(hslab, hbuf + (size_t)(S_ & 1) * HBSZ + (size_t)b0 * H_, tid);
  __syncthreads();
  if (wib == 0) xphase(S_ - 1, true);

#pragma unroll
  for (int v = 0; v < 8; ++v) {
    const int b = b0 + rbase + v;
    out[OUT_CZ + (size_t)b * Z_ + 2 + j0 + l] = creg[v];
  }
}

extern "C" void kernel_launch(void* const* d_in, const int* in_sizes, int n_in,
                              void* d_out, int out_size, void* d_ws, size_t ws_size,
                              hipStream_t stream) {
  (void)in_sizes; (void)n_in; (void)out_size; (void)ws_size;
  const float* rnn_input = (const float*)d_in[0];
  const float* tau       = (const float*)d_in[1];
  const float* z0        = (const float*)d_in[2];
  const float* c0        = (const float*)d_in[3];
  const float* WU_w      = (const float*)d_in[4];
  const float* WU_b      = (const float*)d_in[5];
  const float* alpha_w   = (const float*)d_in[6];
  const float* alpha_b   = (const float*)d_in[7];
  const float* Wh_w      = (const float*)d_in[8];
  const float* Wh_b      = (const float*)d_in[9];
  const float* Amat      = (const float*)d_in[10];
  const float* Bmat      = (const float*)d_in[11];
  unsigned char* ws = (unsigned char*)d_ws;
  float* out = (float*)d_out;

  prep_kernel<<<256, 256, 0, stream>>>(WU_w, z0, ws);
  lstm_persist<<<NBLK, NTHR, 0, stream>>>(rnn_input, tau, z0, c0, WU_b, alpha_w, alpha_b,
                                          Wh_w, Wh_b, Amat, Bmat, out, ws);
}
